// SE3Transformer_26293789786410
// MI455X (gfx1250) — matrix operand order repaired and checked by reading, not compiled, not run on hardware
//
#include <hip/hip_runtime.h>
#include <math.h>

// ---------------- problem constants (from reference setup_inputs) ----------
#define B_    2
#define N_    4096
#define K_    32
#define D0_   64
#define D1_   16
#define L_    2
#define RH_   16
#define NODES (B_*N_)
#define EDGES (NODES*K_)
#define EPS_  1e-6f

typedef __attribute__((ext_vector_type(16))) _Float16 v16h;
typedef __attribute__((ext_vector_type(8)))  float    v8f;

// tanh-approx GELU (jax.nn.gelu default approximate=True)
__device__ __forceinline__ float geluf(float x){
  const float c0=0.7978845608028654f, c1=0.044715f;
  return 0.5f*x*(1.0f+tanhf(c0*(x+c1*x*x*x)));
}

// ---- WMMA fragment loaders (per CDNA5 ISA 7.12.2 layout tables) -----------
// A (16x32 f16): lanes 0-15 rows M=lane, K-halfs {0..7,16..23}; lanes 16-31
// same rows, K-halfs {8..15,24..31}.
__device__ __forceinline__ v16h ldA(const _Float16* base, int stride, int lane){
  int m  = lane & 15;
  int kb = (lane & 16) ? 8 : 0;
  const _Float16* r = base + m*stride + kb;
  v16h a;
#pragma unroll
  for (int j=0;j<8;++j) a[j]   = r[j];
#pragma unroll
  for (int j=0;j<8;++j) a[8+j] = r[16+j];
  return a;
}
// B (32x16 f16): lanes 0-15 column N=lane, K {0..7,16..23}; lanes 16-31
// same columns, K {8..15,24..31}.
__device__ __forceinline__ v16h ldB(const _Float16* base, int stride, int lane){
  int n  = lane & 15;
  int kb = (lane & 16) ? 8 : 0;
  const _Float16* r = base + kb*stride + n;
  v16h b;
#pragma unroll
  for (int j=0;j<8;++j) b[j]   = r[j*stride];
#pragma unroll
  for (int j=0;j<8;++j) b[8+j] = r[(16+j)*stride];
  return b;
}
#define WMMA_F16(A,Bv,C) __builtin_amdgcn_wmma_f32_16x16x32_f16(false,(A),false,(Bv),(short)0,(C),false,false)

// ---------------- small utility kernels ------------------------------------
__global__ void se3_init_state(const float* f0, const float* f1, float* f0s, float* f1s){
  int i = blockIdx.x*256 + threadIdx.x;
  if (i < NODES*D0_) f0s[i] = f0[i];
  int j = i - NODES*D0_;
  if (j >= 0 && j < NODES*D1_*3) f1s[j] = f1[j];
}

__global__ void se3_geom(const float* coords, const int* nbr, float* rhat, float* dist){
  int e = blockIdx.x*256 + threadIdx.x;
  if (e >= EDGES) return;
  int node = e / K_;           // = bi*N_ + ni
  int bi = node / N_, ni = node % N_;
  int idx = nbr[e];
  const float* cs = coords + (size_t)bi*N_*3;
  float rx = cs[idx*3+0]-cs[ni*3+0];
  float ry = cs[idx*3+1]-cs[ni*3+1];
  float rz = cs[idx*3+2]-cs[ni*3+2];
  float d  = sqrtf(rx*rx+ry*ry+rz*rz);
  float inv= 1.0f/(d+EPS_);
  dist[e]=d; rhat[e*3+0]=rx*inv; rhat[e*3+1]=ry*inv; rhat[e*3+2]=rz*inv;
}

// f32 -> f16 weight conversion with optional zero row padding (K-pad to 32)
__global__ void se3_convw(const float* src, _Float16* dst, int srcRows, int dstRows, int cols){
  int i = blockIdx.x*256 + threadIdx.x;
  if (i >= dstRows*cols) return;
  int r = i / cols;
  dst[i] = (r < srcRows) ? (_Float16)src[i] : (_Float16)0.0f;
}

// ---------------- prenorm + Q projection (per node) ------------------------
__global__ __launch_bounds__(64) void se3_prenorm_q(
    const float* f0s, const float* f1s, const float* Wq0, const float* Wq1,
    _Float16* g0h, _Float16* g1h, float* q0, float* q1)
{
  __shared__ float sF0[64], sG0[64], sF1[48], sG1[48];
  int node = blockIdx.x, t = threadIdx.x;
  sF0[t] = f0s[node*64+t];
  if (t < 48) sF1[t] = f1s[node*48+t];
  __syncthreads();
  float mu=0.f;
#pragma unroll
  for (int j=0;j<64;++j) mu += sF0[j];
  mu *= (1.0f/64.0f);
  float var=0.f;
#pragma unroll
  for (int j=0;j<64;++j){ float d=sF0[j]-mu; var += d*d; }
  var *= (1.0f/64.0f);
  float g = (sF0[t]-mu)*rsqrtf(var+EPS_);
  sG0[t]=g; g0h[node*64+t]=(_Float16)g;
  float ss=0.f;
#pragma unroll
  for (int j=0;j<48;++j) ss += sF1[j]*sF1[j];
  float rms = rsqrtf(ss*(1.0f/16.0f)+EPS_);   // mean over d1=16 of per-channel |v|^2
  if (t<48){ float gv=sF1[t]*rms; sG1[t]=gv; g1h[node*48+t]=(_Float16)gv; }
  __syncthreads();
  float acc=0.f;
#pragma unroll
  for (int j=0;j<64;++j) acc += sG0[j]*Wq0[j*64+t];
  q0[node*64+t]=acc;
  if (t<48){
    int E=t/3, m=t%3; float a=0.f;
#pragma unroll
    for (int c=0;c<16;++c) a += sG1[c*3+m]*Wq1[c*16+E];
    q1[node*48+t]=a;
  }
}

// ---------------- fused edge / attention / output kernel -------------------
// One workgroup (128 thr = 4 waves) per node. WMMA for all d0-wide GEMMs.
__global__ __launch_bounds__(128) void se3_edge_attn(
    const _Float16* g0h, const _Float16* g1h, const float* q0g, const float* q1g,
    const float* rhatg, const float* distg, const int* nbr,
    const _Float16* Wk00h, const _Float16* Wv00h,
    const _Float16* Wk10h, const _Float16* Wv10h,   // padded to 32 rows
    const _Float16* Wk01h, const _Float16* Wv01h,
    const float* Wk11, const float* Wk11r, const float* Wv11, const float* Wv11r,
    const float* Rw1, const float* Rb1, const float* Rw20, const float* Rw21,
    const float* Wo0, const float* Wo1,
    float* f0s, float* f1s)
{
  __shared__ int      sIdx[32];
  __shared__ _Float16 sH0[32][72];     // gathered g0 (f16), padded stride
  __shared__ _Float16 sDot[32][40];    // dot1 (f16), K padded 16->32 w/ zeros
  __shared__ float    sH1[32][48];     // gathered g1 (f32), [c*3+m]
  __shared__ float    sDotF[32][16];
  __shared__ float    sRhat[32][3], sDist[32];
  __shared__ float    sR0[32][64], sR1[32][16];
  __shared__ float    sK0[32][64], sV0[32][64];
  __shared__ float    sS01k[32][16], sS01v[32][16];
  __shared__ float    sK1[32][48], sV1[32][48];
  __shared__ float    sLg[32][4], sAt[32][4];
  __shared__ float    sQ0[64], sQ1[48], sO0[64], sO1[48];

  const int node = blockIdx.x;            // bi*N_+ni
  const int bi   = node >> 12;            // N_=4096
  const int tid  = threadIdx.x;
  const int lane = tid & 31, wave = tid >> 5;

  // ---- phase 1: indices, q, gathers, geometry ----
  if (tid < 32) sIdx[tid] = nbr[node*K_ + tid];
  if (tid < 64) sQ0[tid] = q0g[node*64+tid];
  else if (tid < 112) sQ1[tid-64] = q1g[node*48 + tid-64];
  __syncthreads();

  {
    int e = tid >> 2, p = tid & 3;
    int row = bi*N_ + sIdx[e];
    const _Float16* s0 = g0h + (size_t)row*64 + p*16;
#pragma unroll
    for (int j=0;j<16;++j) sH0[e][p*16+j] = s0[j];
    const _Float16* s1 = g1h + (size_t)row*48 + p*12;
#pragma unroll
    for (int j=0;j<12;++j) sH1[e][p*12+j] = (float)s1[j];
  }
  if (tid < 32){
    int eg = node*K_ + tid;
    sRhat[tid][0]=rhatg[eg*3+0]; sRhat[tid][1]=rhatg[eg*3+1]; sRhat[tid][2]=rhatg[eg*3+2];
    sDist[tid]=distg[eg];
  }
  __syncthreads();

  // dot1 + radial MLP (r0, r1)
  if (tid < 32){
#pragma unroll
    for (int c=0;c<16;++c){
      float d = sH1[tid][c*3+0]*sRhat[tid][0] + sH1[tid][c*3+1]*sRhat[tid][1]
              + sH1[tid][c*3+2]*sRhat[tid][2];
      sDotF[tid][c]=d; sDot[tid][c]=(_Float16)d;
    }
#pragma unroll
    for (int c=16;c<32;++c) sDot[tid][c]=(_Float16)0.0f;
  }
  {
    int e = tid >> 2, p = tid & 3;
    float rhv[16]; float dd = sDist[e];
#pragma unroll
    for (int u=0;u<16;++u) rhv[u] = geluf(dd*Rw1[u] + Rb1[u]);
#pragma unroll
    for (int c=0;c<16;++c){
      int cc = p*16+c; float a=0.f;
#pragma unroll
      for (int u=0;u<16;++u) a += rhv[u]*Rw20[u*64+cc];
      sR0[e][cc]=a;
    }
#pragma unroll
    for (int c=0;c<4;++c){
      int cc=p*4+c; float a=0.f;
#pragma unroll
      for (int u=0;u<16;++u) a += rhv[u]*Rw21[u*16+cc];
      sR1[e][cc]=a;
    }
  }
  __syncthreads();

  // ---- phase 2: WMMA GEMMs (all 128 threads active; EXEC all 1s) ----
  {
    // k0/v0: tiles (mat = wave>>1, rowTile = wave&1, colTile = 0..3)
    const int mat = wave >> 1, rt = wave & 1;
    v16h A0 = ldA(&sH0[rt*16][0], 72, lane);
    v16h A1 = ldA(&sH0[rt*16][32], 72, lane);
    v16h Ad = ldA(&sDot[rt*16][0], 40, lane);
    const _Float16* W0 = mat ? Wv00h : Wk00h;
    const _Float16* W1 = mat ? Wv10h : Wk10h;
    float* dst = mat ? &sV0[0][0] : &sK0[0][0];
    const int Nn = lane & 15, Mb = (lane & 16) ? 8 : 0;
#pragma unroll
    for (int ct=0; ct<4; ++ct){
      v16h B0 = ldB(W0 + ct*16,        64, lane);
      v16h B1 = ldB(W0 + 32*64 + ct*16,64, lane);
      v16h Bd = ldB(W1 + ct*16,        64, lane);
      v8f acc = {};
      acc = WMMA_F16(A0, B0, acc);
      acc = WMMA_F16(A1, B1, acc);
      acc = WMMA_F16(Ad, Bd, acc);
#pragma unroll
      for (int r=0;r<8;++r){
        int M = rt*16 + Mb + r, Nc = ct*16 + Nn;
        dst[M*64+Nc] = acc[r]*sR0[M][Nc];   // fold radial scale r0
      }
    }
    // s01 = h0 @ W*01 (64x16): one 16x16 tile per wave
    const _Float16* W = mat ? Wv01h : Wk01h;
    v16h B0 = ldB(W,         16, lane);
    v16h B1 = ldB(W + 32*16, 16, lane);
    v8f acc = {};
    acc = WMMA_F16(A0, B0, acc);
    acc = WMMA_F16(A1, B1, acc);
    float* d2 = mat ? &sS01v[0][0] : &sS01k[0][0];
#pragma unroll
    for (int r=0;r<8;++r) d2[(rt*16+Mb+r)*16 + Nn] = acc[r];
  }
  __syncthreads();

  // ---- phase 3: equivariant k1/v1 (small d1=16 contractions, VALU) ----
  {
    int e = tid >> 2, p = tid & 3;
#pragma unroll
    for (int q=0;q<4;++q){
      int E = p*4+q;
      float sdk=0.f, sdv=0.f;
#pragma unroll
      for (int c=0;c<16;++c){
        float dv = sDotF[e][c];
        sdk += dv*Wk11r[c*16+E]; sdv += dv*Wv11r[c*16+E];
      }
      float bk = sS01k[e][E] + sdk, bv = sS01v[e][E] + sdv;
      float r1v = sR1[e][E];
#pragma unroll
      for (int m=0;m<3;++m){
        float wk=0.f, wv=0.f;
#pragma unroll
        for (int c=0;c<16;++c){
          float hv = sH1[e][c*3+m];
          wk += hv*Wk11[c*16+E]; wv += hv*Wv11[c*16+E];
        }
        float rh = sRhat[e][m];
        sK1[e][E*3+m] = (bk*rh + wk)*r1v;
        sV1[e][E*3+m] = (bv*rh + wv)*r1v;
      }
    }
  }
  __syncthreads();

  // ---- phase 4: logits + softmax over the 32 neighbors ----
  {
    const float scale = rsqrtf((float)(D0_/4 + 3*(D1_/4)));  // 1/sqrt(28)
    int e = tid >> 2, h = tid & 3;
    float lg = 0.f;
#pragma unroll
    for (int c=0;c<16;++c) lg += sQ0[h*16+c]*sK0[e][h*16+c];
#pragma unroll
    for (int c=0;c<4;++c)
#pragma unroll
      for (int m=0;m<3;++m) lg += sQ1[(h*4+c)*3+m]*sK1[e][(h*4+c)*3+m];
    sLg[e][h] = lg*scale;
  }
  __syncthreads();
  if (tid < 4){
    int h = tid; float mx = -3.0e38f;
#pragma unroll
    for (int e=0;e<32;++e) mx = fmaxf(mx, sLg[e][h]);
    float s=0.f;
#pragma unroll
    for (int e=0;e<32;++e){ float ex=expf(sLg[e][h]-mx); sAt[e][h]=ex; s+=ex; }
    float inv=1.0f/s;
#pragma unroll
    for (int e=0;e<32;++e) sAt[e][h]*=inv;
  }
  __syncthreads();

  // ---- phase 5: attention-weighted aggregation ----
  if (tid < 64){
    int c=tid, h=c>>4; float a=0.f;
#pragma unroll
    for (int e=0;e<32;++e) a += sAt[e][h]*sV0[e][c];
    sO0[c]=a;
  } else if (tid < 112){
    int j=tid-64, E=j/3, m=j%3, h=E>>2; float a=0.f;
#pragma unroll
    for (int e=0;e<32;++e) a += sAt[e][h]*sV1[e][E*3+m];
    sO1[j]=a;
  }
  __syncthreads();

  // ---- phase 6: output projection + residual ----
  if (tid < 64){
    int c=tid; float a = f0s[(size_t)node*64+c];
#pragma unroll
    for (int j=0;j<64;++j) a += sO0[j]*Wo0[j*64+c];
    f0s[(size_t)node*64+c]=a;
  } else if (tid < 112){
    int j=tid-64, fo=j/3, m=j%3; float a = f1s[(size_t)node*48+j];
#pragma unroll
    for (int c=0;c<16;++c) a += sO1[c*3+m]*Wo1[c*16+fo];
    f1s[(size_t)node*48+j]=a;
  }
}

// ---------------- FFN (per node) -------------------------------------------
__global__ __launch_bounds__(128) void se3_ffn(
    float* f0s, float* f1s,
    const float* F0w1, const float* F0w2, const float* F1w1, const float* F1w2)
{
  __shared__ float sF0[64], sG0[64], sF1[48], sG1[48], sHid[256], sU1[64*3];
  int node=blockIdx.x, t=threadIdx.x;
  if (t < 64) sF0[t]=f0s[(size_t)node*64+t];
  else if (t < 112) sF1[t-64]=f1s[(size_t)node*48+t-64];
  __syncthreads();
  if (t < 64){
    float mu=0.f;
#pragma unroll
    for (int j=0;j<64;++j) mu += sF0[j];
    mu *= (1.0f/64.0f);
    float var=0.f;
#pragma unroll
    for (int j=0;j<64;++j){ float d=sF0[j]-mu; var+=d*d; }
    var *= (1.0f/64.0f);
    sG0[t]=(sF0[t]-mu)*rsqrtf(var+EPS_);
  } else if (t < 112){
    float ss=0.f;
#pragma unroll
    for (int j=0;j<48;++j) ss += sF1[j]*sF1[j];
    float rms = rsqrtf(ss*(1.0f/16.0f)+EPS_);
    sG1[t-64]=sF1[t-64]*rms;
  }
  __syncthreads();
  // hidden = gelu(g0 @ F0w1), 256 units across 128 threads
#pragma unroll
  for (int u=t; u<256; u+=128){
    float a=0.f;
#pragma unroll
    for (int c=0;c<64;++c) a += sG0[c]*F0w1[c*256+u];
    sHid[u]=geluf(a);
  }
  __syncthreads();
  // u1 = g1 @ F1w1 with sigmoid-norm gate
  if (t < 64){
    float u0=0.f,u1v=0.f,u2=0.f;
#pragma unroll
    for (int c=0;c<16;++c){
      u0 += sG1[c*3+0]*F1w1[c*64+t];
      u1v+= sG1[c*3+1]*F1w1[c*64+t];
      u2 += sG1[c*3+2]*F1w1[c*64+t];
    }
    float nrm = sqrtf(u0*u0+u1v*u1v+u2*u2);
    float gate = 1.0f/(1.0f+expf(-nrm));
    sU1[t*3+0]=u0*gate; sU1[t*3+1]=u1v*gate; sU1[t*3+2]=u2*gate;
  }
  __syncthreads();
  if (t < 64){
    float a=sF0[t];
#pragma unroll
    for (int u=0;u<256;++u) a += sHid[u]*F0w2[u*64+t];
    f0s[(size_t)node*64+t]=a;
  } else if (t < 112){
    int j=t-64, fo=j/3, m=j%3; float a=sF1[j];
#pragma unroll
    for (int E=0;E<64;++E) a += sU1[E*3+m]*F1w2[E*16+fo];
    f1s[(size_t)node*48+j]=a;
  }
}

__global__ __launch_bounds__(128) void se3_writeout(const float* f0s, const float* f1s, float* out){
  int node=blockIdx.x, t=threadIdx.x;
  if (t < 64) out[(size_t)node*112+t]=f0s[(size_t)node*64+t];
  else if (t < 112) out[(size_t)node*112+t]=f1s[(size_t)node*48+t-64];
}

// ---------------- host orchestration ---------------------------------------
extern "C" void kernel_launch(void* const* d_in, const int* in_sizes, int n_in,
                              void* d_out, int out_size, void* d_ws, size_t ws_size,
                              hipStream_t stream)
{
  (void)in_sizes; (void)n_in; (void)out_size; (void)ws_size;
  const float* f0     = (const float*)d_in[0];
  const float* f1     = (const float*)d_in[1];
  const float* coords = (const float*)d_in[2];
  const int*   nbr    = (const int*)  d_in[3];
  const float* Wq0  =(const float*)d_in[4];  const float* Wq1  =(const float*)d_in[5];
  const float* Wk00 =(const float*)d_in[6];  const float* Wk10 =(const float*)d_in[7];
  const float* Wk01 =(const float*)d_in[8];  const float* Wk11 =(const float*)d_in[9];
  const float* Wk11r=(const float*)d_in[10]; const float* Wv00 =(const float*)d_in[11];
  const float* Wv10 =(const float*)d_in[12]; const float* Wv01 =(const float*)d_in[13];
  const float* Wv11 =(const float*)d_in[14]; const float* Wv11r=(const float*)d_in[15];
  const float* Rw1  =(const float*)d_in[16]; const float* Rb1  =(const float*)d_in[17];
  const float* Rw20 =(const float*)d_in[18]; const float* Rw21 =(const float*)d_in[19];
  const float* Wo0  =(const float*)d_in[20]; const float* Wo1  =(const float*)d_in[21];
  const float* F0w1 =(const float*)d_in[22]; const float* F0w2 =(const float*)d_in[23];
  const float* F1w1 =(const float*)d_in[24]; const float* F1w2 =(const float*)d_in[25];

  char* ws = (char*)d_ws; size_t off=0;
  auto alloc=[&](size_t bytes)->char*{ char* p=ws+off; off=(off+bytes+255)&~(size_t)255; return p; };
  float*    f0s  =(float*)   alloc((size_t)NODES*64*4);
  float*    f1s  =(float*)   alloc((size_t)NODES*48*4);
  _Float16* g0h  =(_Float16*)alloc((size_t)NODES*64*2);
  _Float16* g1h  =(_Float16*)alloc((size_t)NODES*48*2);
  float*    q0   =(float*)   alloc((size_t)NODES*64*4);
  float*    q1   =(float*)   alloc((size_t)NODES*48*4);
  float*    rhat =(float*)   alloc((size_t)EDGES*3*4);
  float*    dist =(float*)   alloc((size_t)EDGES*4);
  _Float16* wk00h=(_Float16*)alloc(64*64*2);
  _Float16* wv00h=(_Float16*)alloc(64*64*2);
  _Float16* wk10h=(_Float16*)alloc(32*64*2);   // 16 rows + 16 zero-pad
  _Float16* wv10h=(_Float16*)alloc(32*64*2);
  _Float16* wk01h=(_Float16*)alloc(64*16*2);
  _Float16* wv01h=(_Float16*)alloc(64*16*2);

  se3_init_state<<<(NODES*(64+48)+255)/256,256,0,stream>>>(f0,f1,f0s,f1s);
  se3_geom<<<(EDGES+255)/256,256,0,stream>>>(coords,nbr,rhat,dist);

  for (int l=0;l<L_;++l){
    se3_prenorm_q<<<NODES,64,0,stream>>>(f0s,f1s, Wq0+l*64*64, Wq1+l*16*16, g0h,g1h,q0,q1);
    se3_convw<<<(64*64+255)/256,256,0,stream>>>(Wk00+l*64*64, wk00h, 64,64,64);
    se3_convw<<<(64*64+255)/256,256,0,stream>>>(Wv00+l*64*64, wv00h, 64,64,64);
    se3_convw<<<(32*64+255)/256,256,0,stream>>>(Wk10+l*16*64, wk10h, 16,32,64);
    se3_convw<<<(32*64+255)/256,256,0,stream>>>(Wv10+l*16*64, wv10h, 16,32,64);
    se3_convw<<<(64*16+255)/256,256,0,stream>>>(Wk01+l*64*16, wk01h, 64,64,16);
    se3_convw<<<(64*16+255)/256,256,0,stream>>>(Wv01+l*64*16, wv01h, 64,64,16);
    se3_edge_attn<<<NODES,128,0,stream>>>(
        g0h,g1h,q0,q1,rhat,dist,nbr,
        wk00h,wv00h,wk10h,wv10h,wk01h,wv01h,
        Wk11+l*256, Wk11r+l*256, Wv11+l*256, Wv11r+l*256,
        Rw1+l*16, Rb1+l*16, Rw20+l*16*64, Rw21+l*256,
        Wo0+l*64*64, Wo1+l*256,
        f0s,f1s);
    se3_ffn<<<NODES,128,0,stream>>>(f0s,f1s,
        F0w1+l*64*256, F0w2+l*256*64, F1w1+l*16*64, F1w2+l*64*16);
  }
  se3_writeout<<<NODES,128,0,stream>>>(f0s,f1s,(float*)d_out);
}
